// ReferenceLSTM_71021579207005
// MI455X (gfx1250) — matrix operand order repaired and checked by reading, not compiled, not run on hardware
//
#include <hip/hip_runtime.h>

// ---------------------------------------------------------------------------
// 3-layer LSTM (B=8192, T=336, D=16, H=32) + FC head, fully fused.
// One workgroup = 8 wave32 = one 16-row batch tile across all layers/time.
// Wave w owns gate-column tile w (16 of the 128 gate outputs) and holds its
// six 32x16 f16 B-matrices (W_ih/W_hh for 3 layers) in registers forever.
// Elementwise cell updates use native v_tanh_f32 / v_rcp_f32 (no IEEE divide).
// ---------------------------------------------------------------------------

typedef __attribute__((ext_vector_type(16))) _Float16 v16h;
typedef __attribute__((ext_vector_type(8)))  _Float16 v8h;
typedef __attribute__((ext_vector_type(8)))  float    v8f;
typedef __attribute__((ext_vector_type(4)))  float    v4f;

#define T_STEPS 336
#define IN_D    16
#define HID     32

__device__ __forceinline__ float fast_rcp(float x) {
#if __has_builtin(__builtin_amdgcn_rcpf)
  return __builtin_amdgcn_rcpf(x);   // raw v_rcp_f32, no Newton iteration
#else
  return 1.0f / x;
#endif
}

__device__ __forceinline__ float tanhfast(float x) {
#if __has_builtin(__builtin_amdgcn_tanhf)
  return __builtin_amdgcn_tanhf(x);          // v_tanh_f32 (CDNA5 TRANS op)
#elif __has_builtin(__builtin_amdgcn_tanh_f32)
  return __builtin_amdgcn_tanh_f32(x);
#else
  // tanh(x) = 1 - 2/(1 + e^{2x})
  return 1.0f - 2.0f * fast_rcp(1.0f + __expf(2.0f * x));
#endif
}

__device__ __forceinline__ float sigf(float x) {
#if __has_builtin(__builtin_amdgcn_tanhf) || __has_builtin(__builtin_amdgcn_tanh_f32)
  // sigmoid(x) = 0.5 * tanh(x/2) + 0.5  -> one tanh + one fma
  return __builtin_fmaf(0.5f, tanhfast(0.5f * x), 0.5f);
#else
  return fast_rcp(1.0f + __expf(-x));
#endif
}

__device__ __forceinline__ v8f wmma_f16(v16h a, v16h b, v8f c) {
  // D = A(16x32 f16) * B(32x16 f16) + C(16x16 f32)
  return __builtin_amdgcn_wmma_f32_16x16x32_f16(
      /*neg_a=*/false, a, /*neg_b=*/false, b,
      /*c_mod=*/(short)0, c, /*reuse_a=*/false, /*reuse_b=*/false);
}

// B-matrix layout (32x16 f16, 8 VGPRs): lane = n + 16*kgrp holds halves
// e<8 -> W[n][8*kgrp + e] ; e>=8 -> W[n][16 + 8*kgrp + (e-8)]. W is row-major
// [128][K]. K=16 rows (layer 0 W_ih) are zero-padded to K=32.
__device__ __forceinline__ v16h load_B_row(const float* __restrict__ w,
                                           int K, int n, int kgrp) {
  v16h b;
  {
    const float* p = w + (size_t)n * K + 8 * kgrp;
    v4f a0 = *(const v4f*)(p + 0);
    v4f a1 = *(const v4f*)(p + 4);
    v4f a2 = {0.f, 0.f, 0.f, 0.f};
    v4f a3 = {0.f, 0.f, 0.f, 0.f};
    if (K > 16) {
      a2 = *(const v4f*)(p + 16);
      a3 = *(const v4f*)(p + 20);
    }
#pragma unroll
    for (int i = 0; i < 4; ++i) {
      b[i]      = (_Float16)a0[i];
      b[4 + i]  = (_Float16)a1[i];
      b[8 + i]  = (_Float16)a2[i];
      b[12 + i] = (_Float16)a3[i];
    }
  }
  return b;
}

// A-matrix layout (16x32 f16): lane l (m=l%16, grp=l/16) holds halves
// e<8 -> k=8*grp+e ; e>=8 -> k=16+8*grp+(e-8). Source: row-major f16 [16][32].
__device__ __forceinline__ v16h load_A_h(const _Float16* __restrict__ hbuf,
                                         int m, int grp) {
  v8h lo = *(const v8h*)(hbuf + m * 32 + 8 * grp);
  v8h hi = *(const v8h*)(hbuf + m * 32 + 16 + 8 * grp);
  return __builtin_shufflevector(lo, hi, 0, 1, 2, 3, 4, 5, 6, 7,
                                 8, 9, 10, 11, 12, 13, 14, 15);
}

// A from x (K=16 input, zero-padded to 32). xp -> x[b][t][0], 32B-aligned slice.
__device__ __forceinline__ v16h load_A_x(const float* __restrict__ xp, int grp) {
  const float* p = xp + 8 * grp;
  v4f a0 = *(const v4f*)(p + 0);
  v4f a1 = *(const v4f*)(p + 4);
  v16h r;
#pragma unroll
  for (int i = 0; i < 4; ++i) {
    r[i]      = (_Float16)a0[i];
    r[4 + i]  = (_Float16)a1[i];
    r[8 + i]  = (_Float16)0.0f;   // k = 16..31 padded (B rows there are 0 too)
    r[12 + i] = (_Float16)0.0f;
  }
  return r;
}

// C layout: VGPR v, lane l -> (m = v + 8*(l/16), n = l%16). Store gates as
// G[n][m] so each lane writes 8 contiguous floats (two b128 stores).
__device__ __forceinline__ void store_gates(float (*G)[16], v8f acc,
                                            int n, int grp) {
  v4f lo = __builtin_shufflevector(acc, acc, 0, 1, 2, 3);
  v4f hi = __builtin_shufflevector(acc, acc, 4, 5, 6, 7);
  *(v4f*)&G[n][8 * grp]     = lo;
  *(v4f*)&G[n][8 * grp + 4] = hi;
}

__global__ __launch_bounds__(256) void lstm3_wmma_kernel(
    const float* __restrict__ x,
    const float* __restrict__ wih0, const float* __restrict__ whh0,
    const float* __restrict__ bih0, const float* __restrict__ bhh0,
    const float* __restrict__ wih1, const float* __restrict__ whh1,
    const float* __restrict__ bih1, const float* __restrict__ bhh1,
    const float* __restrict__ wih2, const float* __restrict__ whh2,
    const float* __restrict__ bih2, const float* __restrict__ bhh2,
    const float* __restrict__ fc1w, const float* __restrict__ fc1b,
    const float* __restrict__ fc2w, const float* __restrict__ fc2b,
    float* __restrict__ out) {
  __shared__ _Float16 hA[3][16][32];  // layer hidden state, row-major f16
  __shared__ float    G[128][16];     // gates [gate][batch-row], shared by layers

  const int tid  = threadIdx.x;
  const int wv   = tid >> 5;                 // wave = gate N-tile 0..7
  const int lane = tid & 31;
  const int nl   = lane & 15;
  const int grp  = lane >> 4;
  const int base = blockIdx.x * 16;          // batch tile start

  // zero-init hidden state
  for (int i = tid; i < 3 * 16 * 32; i += 256)
    ((_Float16*)hA)[i] = (_Float16)0.0f;

  // persistent per-wave weights (B layout) + per-lane gate bias
  const int ng = 16 * wv + nl;  // this lane's gate row 0..127
  v16h Bih0 = load_B_row(wih0, 16, ng, grp);
  v16h Bhh0 = load_B_row(whh0, 32, ng, grp);
  v16h Bih1 = load_B_row(wih1, 32, ng, grp);
  v16h Bhh1 = load_B_row(whh1, 32, ng, grp);
  v16h Bih2 = load_B_row(wih2, 32, ng, grp);
  v16h Bhh2 = load_B_row(whh2, 32, ng, grp);
  const float bias0 = bih0[ng] + bhh0[ng];
  const float bias1 = bih1[ng] + bhh1[ng];
  const float bias2 = bih2[ng] + bhh2[ng];

  // elementwise assignment: this lane owns (row em, cols ej and ej+16)
  const int em = 2 * wv + grp;
  const int ej = nl;
  float c0a = 0.f, c0b = 0.f, c1a = 0.f, c1b = 0.f, c2a = 0.f, c2b = 0.f;

  const int m = nl;  // A-matrix row for this lane

  __syncthreads();

  for (int t = 0; t < T_STEPS; ++t) {
#pragma unroll
    for (int l = 0; l < 3; ++l) {
      // ---- gates = [in | h] x [W_ih ; W_hh]^T + b  (two K=32 WMMAs) ----
      v16h Ain;
      if (l == 0) {
        const float* xp = x + ((size_t)(base + m) * T_STEPS + t) * IN_D;
        Ain = load_A_x(xp, grp);
      } else {
        Ain = load_A_h(&hA[l - 1][0][0], m, grp);
      }
      const float bias = (l == 0) ? bias0 : (l == 1) ? bias1 : bias2;
      v8f acc = {bias, bias, bias, bias, bias, bias, bias, bias};
      acc = wmma_f16(Ain, (l == 0) ? Bih0 : (l == 1) ? Bih1 : Bih2, acc);
      v16h Ah = load_A_h(&hA[l][0][0], m, grp);
      acc = wmma_f16(Ah, (l == 0) ? Bhh0 : (l == 1) ? Bhh1 : Bhh2, acc);
      store_gates(G, acc, ng, grp);
      __syncthreads();

      // ---- elementwise LSTM cell update (PyTorch gate order i,f,g,o) ----
      float& cA = (l == 0) ? c0a : (l == 1) ? c1a : c2a;
      float& cB = (l == 0) ? c0b : (l == 1) ? c1b : c2b;
      {
        float gi = G[ej][em], gf = G[32 + ej][em];
        float gg = G[64 + ej][em], go = G[96 + ej][em];
        cA = sigf(gf) * cA + sigf(gi) * tanhfast(gg);
        hA[l][em][ej] = (_Float16)(sigf(go) * tanhfast(cA));
      }
      {
        const int j2 = ej + 16;
        float gi = G[j2][em], gf = G[32 + j2][em];
        float gg = G[64 + j2][em], go = G[96 + j2][em];
        cB = sigf(gf) * cB + sigf(gi) * tanhfast(gg);
        hA[l][em][j2] = (_Float16)(sigf(go) * tanhfast(cB));
      }
      __syncthreads();
    }
  }

  // ---- FC head on last hidden state of layer 2 (tiny: 16 lanes do it) ----
  if (wv == 0 && lane < 16) {
    const int mm = lane;
    float o = fc2b[0];
#pragma unroll
    for (int n = 0; n < 16; ++n) {
      float s = fc1b[n];
#pragma unroll
      for (int k = 0; k < HID; ++k)
        s += (float)hA[2][mm][k] * fc1w[n * HID + k];
      s = fmaxf(s, 0.0f);
      o += s * fc2w[n];
    }
    out[base + mm] = o;
  }
}

extern "C" void kernel_launch(void* const* d_in, const int* in_sizes, int n_in,
                              void* d_out, int out_size, void* d_ws,
                              size_t ws_size, hipStream_t stream) {
  (void)in_sizes; (void)n_in; (void)out_size; (void)d_ws; (void)ws_size;
  const float* x    = (const float*)d_in[0];
  const float* wih0 = (const float*)d_in[1];
  const float* whh0 = (const float*)d_in[2];
  const float* bih0 = (const float*)d_in[3];
  const float* bhh0 = (const float*)d_in[4];
  const float* wih1 = (const float*)d_in[5];
  const float* whh1 = (const float*)d_in[6];
  const float* bih1 = (const float*)d_in[7];
  const float* bhh1 = (const float*)d_in[8];
  const float* wih2 = (const float*)d_in[9];
  const float* whh2 = (const float*)d_in[10];
  const float* bih2 = (const float*)d_in[11];
  const float* bhh2 = (const float*)d_in[12];
  const float* fc1w = (const float*)d_in[13];
  const float* fc1b = (const float*)d_in[14];
  const float* fc2w = (const float*)d_in[15];
  const float* fc2b = (const float*)d_in[16];
  float* out = (float*)d_out;

  const int B = 8192;
  dim3 grid(B / 16), block(256);
  lstm3_wmma_kernel<<<grid, block, 0, stream>>>(
      x, wih0, whh0, bih0, bhh0, wih1, whh1, bih1, bhh1,
      wih2, whh2, bih2, bhh2, fc1w, fc1b, fc2w, fc2b, out);
}
